// PerformerAttention_38783554683384
// MI455X (gfx1250) — matrix operand order repaired and checked by reading, not compiled, not run on hardware
//
#include <hip/hip_runtime.h>

typedef __attribute__((ext_vector_type(16))) __bf16 v16bf;
typedef __attribute__((ext_vector_type(8)))  float  v8f;
typedef __attribute__((ext_vector_type(4)))  int    v4i;

#define B_  4
#define H_  16
#define N_  4096
#define D_  64
#define M_  256
#define BH_ 64

static constexpr float DATA_NORM  = 0.35355339059327373f; // 64^-0.25
static constexpr float DIAG_SCALE = 0.0625f;              // 0.5 * 64^-0.5
static constexpr float RATIO      = 0.0625f;              // 256^-0.5
static constexpr float EPS_       = 1e-4f;

#if defined(__has_builtin)
#if __has_builtin(__builtin_amdgcn_global_load_async_to_lds_b128)
#define HAVE_ASYNC_LDS 1
#endif
#endif

#ifdef HAVE_ASYNC_LDS
// builtin prototype: (v4i addrspace(1)* src, v4i addrspace(3)* dst, imm, imm)
#define ASYNC_B128(gsrc, ldst)                                                 \
  __builtin_amdgcn_global_load_async_to_lds_b128(                              \
      (__attribute__((address_space(1))) v4i*)(gsrc),                          \
      (__attribute__((address_space(3))) v4i*)(ldst), 0, 0)
#endif

// ---------------------------------------------------------------------------
// WMMA helpers (bf16 in, f32 accumulate)
// ---------------------------------------------------------------------------
__device__ __forceinline__ v8f wmma_bf16(v16bf a, v16bf b, v8f c) {
  return __builtin_amdgcn_wmma_f32_16x16x32_bf16(false, a, false, b,
                                                 (short)0, c, false, false);
}

// A operand (16x32, 16-bit): lane 0-15 -> rows, per-lane 16 consecutive K
// starting at half*16. Loads row of X (row-major, stride D_) scaled, f32->bf16.
__device__ __forceinline__ v16bf a_from_global(const float* x, int row, int k0,
                                               float scale) {
  int lane = threadIdx.x & 31;
  int half = lane >> 4;
  const float* p = x + (size_t)row * D_ + k0 + half * 16;
  v16bf a;
#pragma unroll
  for (int j = 0; j < 8; ++j) a[j] = (__bf16)(scale * p[j]);
#pragma unroll
  for (int j = 0; j < 8; ++j) a[8 + j] = (__bf16)(scale * p[8 + j]);
  return a;
}

// B operand (32x16, 16-bit): lane&15 -> column m, per-lane 16 consecutive K
// starting at half*16. proj_bf16 is [M_][D_] row-major, B[k][m] = proj[m][k].
__device__ __forceinline__ v16bf b_from_proj(const __bf16* projb, int m0, int k0) {
  int lane = threadIdx.x & 31;
  const __bf16* p = projb + (size_t)(m0 + (lane & 15)) * D_ + k0 + (lane >> 4) * 16;
  v16bf b;
#pragma unroll
  for (int j = 0; j < 16; ++j) b[j] = p[j];
  return b;
}

// ---------------------------------------------------------------------------
// K0: zero accumulators, convert projection matrix to bf16
// ---------------------------------------------------------------------------
__global__ __launch_bounds__(256) void k_init(float* ctx, float* kc,
                                              __bf16* projb, const float* proj) {
  int idx = blockIdx.x * 256 + threadIdx.x;
  if (idx < BH_ * M_ * D_) ctx[idx] = 0.0f;
  if (idx < BH_ * M_)      kc[idx]  = 0.0f;
  if (idx < M_ * D_)       projb[idx] = (__bf16)proj[idx];
}

// ---------------------------------------------------------------------------
// K1: global max of data_dash for keys (per-workgroup partial max)
// grid = BH_ * 16 (256 rows per WG, 32 per wave)
// ---------------------------------------------------------------------------
__global__ __launch_bounds__(256) void k_kmax(const float* k, const __bf16* projb,
                                              float* partial) {
  __shared__ float wmax[8];
  int wg   = blockIdx.x;
  int bh   = wg >> 4;
  int nc   = wg & 15;
  int wave = threadIdx.x >> 5;
  int lane = threadIdx.x & 31;
  const float* kb = k + (size_t)bh * N_ * D_;
  int row0 = nc * 256 + wave * 32;

  float mx = -3.4e38f;
  for (int a = 0; a < 2; ++a) {
    int r = row0 + a * 16 + (lane & 15);
    v16bf a0 = a_from_global(kb, r, 0,  DATA_NORM);
    v16bf a1 = a_from_global(kb, r, 32, DATA_NORM);
    for (int mt = 0; mt < 16; ++mt) {
      v8f dd = {};
      dd = wmma_bf16(a0, b_from_proj(projb, mt * 16, 0),  dd);
      dd = wmma_bf16(a1, b_from_proj(projb, mt * 16, 32), dd);
#pragma unroll
      for (int i = 0; i < 8; ++i) mx = fmaxf(mx, dd[i]);
    }
  }
#pragma unroll
  for (int s = 16; s >= 1; s >>= 1) mx = fmaxf(mx, __shfl_xor(mx, s, 32));
  if (lane == 0) wmax[wave] = mx;
  __syncthreads();
  if (threadIdx.x == 0) {
    float m = wmax[0];
    for (int i = 1; i < 8; ++i) m = fmaxf(m, wmax[i]);
    partial[wg] = m;
  }
}

__global__ __launch_bounds__(256) void k_kmax_reduce(const float* partial,
                                                     float* gmax) {
  __shared__ float wmax[8];
  int t = threadIdx.x;
  float m = fmaxf(fmaxf(partial[t], partial[t + 256]),
                  fmaxf(partial[t + 512], partial[t + 768]));
#pragma unroll
  for (int s = 16; s >= 1; s >>= 1) m = fmaxf(m, __shfl_xor(m, s, 32));
  if ((t & 31) == 0) wmax[t >> 5] = m;
  __syncthreads();
  if (t == 0) {
    float r = wmax[0];
    for (int i = 1; i < 8; ++i) r = fmaxf(r, wmax[i]);
    *gmax = r;
  }
}

// ---------------------------------------------------------------------------
// K2: kp = ratio*(exp(k.P^T - diag - MX)+eps); k_cumsum += sum_n kp;
//     context += kp^T @ v.   grid = BH_ * 4 (1024 rows per WG in 32-row chunks)
// wave w owns context m-tiles {2w,2w+1} x all 4 e-tiles.
// v chunk staged via async global->LDS copy (ASYNCcnt), k/v prefetched.
// ---------------------------------------------------------------------------
__global__ __launch_bounds__(256) void k_ctx(const float* k, const float* v,
                                             const __bf16* projb,
                                             const float* gmaxp,
                                             float* ctx, float* kc) {
  __shared__ float  v_lds32[32 * 64];     // v chunk, f32 (8 KB)
  __shared__ float  diag_lds[32];
  __shared__ float  kc_lds[M_];
  __shared__ __bf16 kpT[8][32 * 32];      // per-wave kp^T block [m_local][n_local]

  int wg   = blockIdx.x;
  int bh   = wg >> 2;
  int qtr  = wg & 3;
  int wave = threadIdx.x >> 5;
  int lane = threadIdx.x & 31;
  int half = lane >> 4;
  int tid  = threadIdx.x;
  const float* kb = k + (size_t)bh * N_ * D_;
  const float* vb = v + (size_t)bh * N_ * D_;
  float MX = *gmaxp;

  kc_lds[tid] = 0.0f;

  v8f acc[2][4] = {};   // context accumulators

  int nbase = qtr * 1024;
  for (int c = 0; c < 32; ++c) {
    int n0 = nbase + c * 32;
    __syncthreads();                     // protect v_lds / diag_lds reuse

    // prefetch next chunk's k and v rows (global_prefetch)
    if (c < 31) {
      __builtin_prefetch(kb + (size_t)(n0 + 32) * D_ + tid * 8, 0, 0);
      __builtin_prefetch(vb + (size_t)(n0 + 32) * D_ + tid * 8, 0, 0);
    }

    // stage v chunk [32][64] f32 into LDS
#ifdef HAVE_ASYNC_LDS
    {
      const float* src = vb + (size_t)n0 * D_;
#pragma unroll
      for (int it = 0; it < 2; ++it) {
        int idx = (it * 256 + tid) * 4;           // 16B per lane
        ASYNC_B128(src + idx, &v_lds32[idx]);
      }
      asm volatile("s_wait_asynccnt 0" ::: "memory");
    }
#else
    for (int i = tid; i < 32 * 64; i += 256)
      v_lds32[i] = vb[(size_t)n0 * D_ + i];
#endif
    if (tid < 32) {
      const float* kr = kb + (size_t)(n0 + tid) * D_;
      float s = 0.0f;
      for (int d = 0; d < D_; ++d) { float x = kr[d]; s += x * x; }
      diag_lds[tid] = DIAG_SCALE * s;
    }
    __syncthreads();

    // dd for m-tiles {2w,2w+1}, n-tiles a=0,1; exp -> kpT block + k_cumsum
    for (int a = 0; a < 2; ++a) {
      int r = n0 + a * 16 + (lane & 15);
      v16bf a0 = a_from_global(kb, r, 0,  DATA_NORM);
      v16bf a1 = a_from_global(kb, r, 32, DATA_NORM);
      for (int t = 0; t < 2; ++t) {
        int mt = wave * 2 + t;
        v8f dd = {};
        dd = wmma_bf16(a0, b_from_proj(projb, mt * 16, 0),  dd);
        dd = wmma_bf16(a1, b_from_proj(projb, mt * 16, 32), dd);
        float csum = 0.0f;
#pragma unroll
        for (int i = 0; i < 8; ++i) {
          int rloc = a * 16 + i + 8 * half;          // chunk-local row
          float kpv = RATIO * (__expf(dd[i] - diag_lds[rloc] - MX) + EPS_);
          csum += kpv;
          kpT[wave][(t * 16 + (lane & 15)) * 32 + rloc] = (__bf16)kpv;
        }
        csum += __shfl_xor(csum, 16, 32);            // add other row-half
        if (half == 0) atomicAdd(&kc_lds[mt * 16 + (lane & 15)], csum);
      }
    }
    asm volatile("s_wait_dscnt 0" ::: "memory");     // in-wave LDS RAW

    // context GEMM: A = kpT block, B = v chunk (K = 32 rows of n)
    v16bf bv[4];
#pragma unroll
    for (int e = 0; e < 4; ++e) {
      int ecol = e * 16 + (lane & 15);
#pragma unroll
      for (int j = 0; j < 16; ++j)
        bv[e][j] = (__bf16)v_lds32[(half * 16 + j) * 64 + ecol];
    }
    for (int t = 0; t < 2; ++t) {
      v16bf am;
      const __bf16* p = &kpT[wave][(t * 16 + (lane & 15)) * 32 + half * 16];
#pragma unroll
      for (int j = 0; j < 8; ++j) am[j] = p[j];
#pragma unroll
      for (int j = 0; j < 8; ++j) am[8 + j] = p[8 + j];
      for (int e = 0; e < 4; ++e) acc[t][e] = wmma_bf16(am, bv[e], acc[t][e]);
    }
  }
  __syncthreads();

  if (tid < M_) atomicAdd(&kc[bh * M_ + tid], kc_lds[tid]);

  float* cb = ctx + (size_t)bh * M_ * D_;
  for (int t = 0; t < 2; ++t)
    for (int e = 0; e < 4; ++e)
#pragma unroll
      for (int i = 0; i < 8; ++i) {
        int m    = (wave * 2 + t) * 16 + i + 8 * half;
        int ecol = e * 16 + (lane & 15);
        atomicAdd(&cb[m * D_ + ecol], acc[t][e][i]);
      }
}

// ---------------------------------------------------------------------------
// K2b: convert accumulated context f32 -> bf16 (so K3 can async-copy raw bytes)
// ---------------------------------------------------------------------------
__global__ __launch_bounds__(256) void k_cvt(const float* ctx, __bf16* ctxb) {
  int idx = blockIdx.x * 256 + threadIdx.x;
  if (idx < BH_ * M_ * D_) ctxb[idx] = (__bf16)ctx[idx];
}

// ---------------------------------------------------------------------------
// K3: qp (row-max softmax kernel), denom = qp.k_cumsum, out = (qp@ctx)/denom
// grid = BH_ * 16 (256 rows per WG, each wave owns 32 rows independently)
// context (bf16, 32 KB) staged via async global->LDS copy.
// ---------------------------------------------------------------------------
__global__ __launch_bounds__(256) void k_out(const float* q, const __bf16* projb,
                                             const __bf16* ctxb, const float* kc,
                                             float* out) {
  __shared__ __bf16 ctx_lds[M_ * D_];     // 32 KB, bf16 context
  __shared__ float  kc_lds[M_];
  __shared__ float  diag_lds[256];
  __shared__ __bf16 qp_blk[8][32 * 32];   // per-wave qp block [row][m_local]

  int wg   = blockIdx.x;
  int bh   = wg >> 4;
  int nc   = wg & 15;
  int wave = threadIdx.x >> 5;
  int lane = threadIdx.x & 31;
  int half = lane >> 4;
  int tid  = threadIdx.x;
  const float*  qb = q + (size_t)bh * N_ * D_;
  const __bf16* cb = ctxb + (size_t)bh * M_ * D_;
  int nwg0 = nc * 256;

#ifdef HAVE_ASYNC_LDS
  {
#pragma unroll
    for (int it = 0; it < 8; ++it) {
      int idx = (it * 256 + tid) * 8;               // 8 bf16 = 16B per lane
      ASYNC_B128(cb + idx, &ctx_lds[idx]);
    }
    asm volatile("s_wait_asynccnt 0" ::: "memory");
  }
#else
  for (int i = tid; i < M_ * D_; i += 256) ctx_lds[i] = cb[i];
#endif
  kc_lds[tid] = kc[bh * M_ + tid];
  {
    const float* qr = qb + (size_t)(nwg0 + tid) * D_;
    float s = 0.0f;
    for (int d = 0; d < D_; ++d) { float x = qr[d]; s += x * x; }
    diag_lds[tid] = DIAG_SCALE * s;
  }
  __syncthreads();

  int row0 = nwg0 + wave * 32;
  v16bf aA[2][2];
  for (int a = 0; a < 2; ++a) {
    int r = row0 + a * 16 + (lane & 15);
    aA[a][0] = a_from_global(qb, r, 0,  DATA_NORM);
    aA[a][1] = a_from_global(qb, r, 32, DATA_NORM);
  }

  // pass 1: per-row max of data_dash over all M
  float rmax[2][8];
#pragma unroll
  for (int a = 0; a < 2; ++a)
#pragma unroll
    for (int i = 0; i < 8; ++i) rmax[a][i] = -3.4e38f;
  for (int mt = 0; mt < 16; ++mt) {
    v16bf b0 = b_from_proj(projb, mt * 16, 0);
    v16bf b1 = b_from_proj(projb, mt * 16, 32);
    for (int a = 0; a < 2; ++a) {
      v8f dd = {};
      dd = wmma_bf16(aA[a][0], b0, dd);
      dd = wmma_bf16(aA[a][1], b1, dd);
#pragma unroll
      for (int i = 0; i < 8; ++i) rmax[a][i] = fmaxf(rmax[a][i], dd[i]);
    }
  }
#pragma unroll
  for (int a = 0; a < 2; ++a)
#pragma unroll
    for (int i = 0; i < 8; ++i) {
      float m = rmax[a][i];
#pragma unroll
      for (int s = 8; s >= 1; s >>= 1) m = fmaxf(m, __shfl_xor(m, s, 32));
      rmax[a][i] = m;                    // full row max (within lane half)
    }

  // pass 2: recompute dd, exp -> qp, denom, out-GEMM against context
  v8f oacc[2][4] = {};
  float dsum[2][8] = {};

  for (int mc = 0; mc < 8; ++mc) {       // m chunks of 32
    for (int t = 0; t < 2; ++t) {
      int mt = mc * 2 + t;
      v16bf b0 = b_from_proj(projb, mt * 16, 0);
      v16bf b1 = b_from_proj(projb, mt * 16, 32);
      float kcv = kc_lds[mt * 16 + (lane & 15)];
      for (int a = 0; a < 2; ++a) {
        v8f dd = {};
        dd = wmma_bf16(aA[a][0], b0, dd);
        dd = wmma_bf16(aA[a][1], b1, dd);
#pragma unroll
        for (int i = 0; i < 8; ++i) {
          int rloc = a * 16 + i + 8 * half;
          float qp = RATIO *
              (__expf(dd[i] - diag_lds[wave * 32 + rloc] - rmax[a][i]) + EPS_);
          dsum[a][i] += qp * kcv;
          qp_blk[wave][rloc * 32 + (t * 16 + (lane & 15))] = (__bf16)qp;
        }
      }
    }
    asm volatile("s_wait_dscnt 0" ::: "memory");

    int m0 = mc * 32;
    v16bf bv[4];
#pragma unroll
    for (int e = 0; e < 4; ++e) {
      int ecol = e * 16 + (lane & 15);
#pragma unroll
      for (int j = 0; j < 16; ++j)
        bv[e][j] = ctx_lds[(m0 + half * 16 + j) * D_ + ecol];
    }
    for (int a = 0; a < 2; ++a) {
      v16bf am;
      const __bf16* p = &qp_blk[wave][(a * 16 + (lane & 15)) * 32 + half * 16];
#pragma unroll
      for (int j = 0; j < 8; ++j) am[j] = p[j];
#pragma unroll
      for (int j = 0; j < 8; ++j) am[8 + j] = p[8 + j];
      for (int e = 0; e < 4; ++e) oacc[a][e] = wmma_bf16(am, bv[e], oacc[a][e]);
    }
  }

  // reduce denominators across the 16-lane half, then scale + store
#pragma unroll
  for (int a = 0; a < 2; ++a)
#pragma unroll
    for (int i = 0; i < 8; ++i) {
      float s = dsum[a][i];
#pragma unroll
      for (int sh = 8; sh >= 1; sh >>= 1) s += __shfl_xor(s, sh, 32);
      dsum[a][i] = 1.0f / s;
    }

  float* ob = out + (size_t)bh * N_ * D_;
  for (int a = 0; a < 2; ++a)
    for (int e = 0; e < 4; ++e)
#pragma unroll
      for (int i = 0; i < 8; ++i) {
        int row  = row0 + a * 16 + i + 8 * half;
        int ecol = e * 16 + (lane & 15);
        ob[(size_t)row * D_ + ecol] = oacc[a][e][i] * dsum[a][i];
      }
}

// ---------------------------------------------------------------------------
extern "C" void kernel_launch(void* const* d_in, const int* in_sizes, int n_in,
                              void* d_out, int out_size, void* d_ws, size_t ws_size,
                              hipStream_t stream) {
  const float* q    = (const float*)d_in[0];
  const float* k    = (const float*)d_in[1];
  const float* v    = (const float*)d_in[2];
  const float* proj = (const float*)d_in[3];
  float* out = (float*)d_out;

  float*  ctx     = (float*)d_ws;                       // BH*M*D f32 (4 MB)
  float*  kc      = ctx + (size_t)BH_ * M_ * D_;        // BH*M f32
  float*  partial = kc + (size_t)BH_ * M_;              // 1024 f32
  float*  gmax    = partial + 1024;                     // 1 f32
  __bf16* projb   = (__bf16*)(gmax + 16);               // M*D bf16 (32B aligned)
  __bf16* ctxb    = projb + (size_t)M_ * D_;            // BH*M*D bf16 (2 MB)

  k_init<<<(BH_ * M_ * D_ + 255) / 256, 256, 0, stream>>>(ctx, kc, projb, proj);
  k_kmax<<<BH_ * 16, 256, 0, stream>>>(k, projb, partial);
  k_kmax_reduce<<<1, 256, 0, stream>>>(partial, gmax);
  k_ctx<<<BH_ * 4, 256, 0, stream>>>(k, v, projb, gmax, ctx, kc);
  k_cvt<<<(BH_ * M_ * D_ + 255) / 256, 256, 0, stream>>>(ctx, ctxb);
  k_out<<<BH_ * 16, 256, 0, stream>>>(q, projb, ctxb, kc, out);
}
